// EffNetAttention_1288490188888
// MI455X (gfx1250) — matrix operand order repaired and checked by reading, not compiled, not run on hardware
//
#include <hip/hip_runtime.h>
#include <hip/hip_bf16.h>
#include <stdint.h>

// ---------------------------------------------------------------------------
// CDNA5 / gfx1250 LSTM stack (EffNetAttention reference)
//   B=64, T=1056, D=H=128, gates G=512, DROP=0.75 -> total_len=792
// GEMMs: v_wmma_f32_16x16x32_bf16 (f32 accumulate).
// Recurrence: single persistent workgroup, WGP-resident state,
//             xg seed staged via global_load_async_to_lds_b128 (ASYNCcnt),
//             overlapped with the WMMA phase of the next step.
// ---------------------------------------------------------------------------

typedef __attribute__((ext_vector_type(16))) __bf16 v16bf;
typedef __attribute__((ext_vector_type(8)))  float  v8f;

#define Bn 64
#define Tn 1056
#define Hn 128
#define Gn 512
#define TOTAL_LEN 792

__device__ __forceinline__ float fast_sigmoid(float x) {
    return 1.0f / (1.0f + __expf(-x));
}
__device__ __forceinline__ float fast_tanh(float x) {
    x = fminf(15.0f, fmaxf(-15.0f, x));
    float e = __expf(2.0f * x);
    return (e - 1.0f) / (e + 1.0f);
}

// ---------------------------------------------------------------------------
// Kernel 1: xg = X @ Wih^T + bih + bhh      X:[N,128] f32, Wih:[512,128] f32
// One 16x16 output tile per wave, K=128 -> 4 chained bf16 WMMAs.
// ---------------------------------------------------------------------------
__global__ __launch_bounds__(256) void gemm_xg_kernel(
    const float* __restrict__ X, const float* __restrict__ Wih,
    const float* __restrict__ bih, const float* __restrict__ bhh,
    float* __restrict__ xg, int Nrows)
{
    const int lane = threadIdx.x & 31;
    const int w    = threadIdx.x >> 5;
    const int tile = blockIdx.x * 8 + w;
    const int rowTiles = Nrows >> 4;
    const int rt = tile >> 5;   // 32 column tiles (512/16)
    const int ct = tile & 31;
    if (rt >= rowTiles) return; // wave-uniform

    const int half = lane >> 4;
    const int l15  = lane & 15;
    const int m = rt * 16 + l15;   // A row
    const int n = ct * 16 + l15;   // B row (N index)

    v16bf A[4], Bf[4];
#pragma unroll
    for (int kc = 0; kc < 4; ++kc) {
        // A layout: lane-half selects contiguous K block of 16
        const float* ax = X + (size_t)m * Hn + kc * 32 + half * 16;
        // B layout: lane-half selects contiguous K block of 16
        const float* bx = Wih + (size_t)n * Hn + kc * 32 + half * 16;
        v16bf av, bv;
#pragma unroll
        for (int e = 0; e < 8; ++e) {
            av[e]     = (__bf16)ax[e];
            av[8 + e] = (__bf16)ax[8 + e];
        }
#pragma unroll
        for (int e = 0; e < 16; ++e) bv[e] = (__bf16)bx[e];
        A[kc] = av; Bf[kc] = bv;
    }

    const float bias = bih[n] + bhh[n];
    v8f acc;
#pragma unroll
    for (int e = 0; e < 8; ++e) acc[e] = bias;

#pragma unroll
    for (int kc = 0; kc < 4; ++kc)
        acc = __builtin_amdgcn_wmma_f32_16x16x32_bf16(
            false, A[kc], false, Bf[kc], (short)0, acc, false, false);

    // C/D layout: col = lane&15, row = vgpr + 8*(lane>>4)
#pragma unroll
    for (int e = 0; e < 8; ++e) {
        const int row = rt * 16 + half * 8 + e;
        xg[(size_t)row * Gn + n] = acc[e];
    }
}

// ---------------------------------------------------------------------------
// Kernel 2: persistent single-workgroup LSTM recurrence for one layer.
//   32 waves; wave w owns m-block (w&3) and 4 n-tiles ((w>>2)*4 ..).
//   Whh bf16 fragments register-resident (4x4 v16bf = 128 VGPRs/lane).
//   LDS: xg stage f32 [64][512] | raw gates f32 [64][512]
//        | c f32 [64][128] | h bf16 [64][128]   => 304 KB (<320 KB/WGP)
// ---------------------------------------------------------------------------
__global__ __launch_bounds__(1024, 1) void lstm_recurrent_kernel(
    const float* __restrict__ xg,   // [B*T, 512], row = b*T + t
    const float* __restrict__ Whh,  // [512, 128] f32
    float* __restrict__ hout,       // [B*T, 128] f32, row = b*T + t
    int T)
{
    extern __shared__ char smem[];
    float*  xstage = (float*)smem;                                   // 128 KB
    float*  gates  = (float*)(smem + (size_t)Bn * Gn * 4);           // 128 KB
    float*  cstate = (float*)(smem + (size_t)Bn * Gn * 8);           //  32 KB
    __bf16* hstate = (__bf16*)(smem + (size_t)Bn * Gn * 8
                                    + (size_t)Bn * Hn * 4);          //  16 KB

    const int tid  = threadIdx.x;
    const int lane = tid & 31;
    const int w    = tid >> 5;
    const int mb   = w & 3;   // batch-row block: rows mb*16 .. mb*16+15
    const int ng   = w >> 2;  // n-tile group: tiles ng*4 .. ng*4+3
    const int half = lane >> 4;
    const int l15  = lane & 15;

    // ---- load Whh once into register-resident bf16 B fragments ----
    v16bf Bfrag[4][4];
#pragma unroll
    for (int j = 0; j < 4; ++j) {
        const int n = (ng * 4 + j) * 16 + l15;
#pragma unroll
        for (int kc = 0; kc < 4; ++kc) {
            const float* src = Whh + (size_t)n * Hn + kc * 32 + half * 16;
            v16bf bv;
#pragma unroll
            for (int e = 0; e < 16; ++e) bv[e] = (__bf16)src[e];
            Bfrag[j][kc] = bv;
        }
    }

    // ---- zero h, c state ----
    for (int i = tid; i < Bn * Hn; i += 1024) {
        cstate[i] = 0.0f;
        hstate[i] = (__bf16)0.0f;
    }

    // ---- async stage helper: copy xg[:, t, :] (64x512 f32) into xstage ----
    // Thread tid copies one contiguous 128 B run: elements [tid*32, tid*32+32)
    // of the [b][g] plane (runs never cross a batch row since 512 % 32 == 0).
    const int cb  = (tid * 32) >> 9;   // batch row of this thread's run
    const int cg0 = (tid * 32) & 511;  // starting gate column
    const uint32_t lbase =
        (uint32_t)(uintptr_t)(xstage + tid * 32); // LDS byte address (aperture low bits)

    auto issue_stage = [&](int t) {
        uint64_t gaddr = (uint64_t)(uintptr_t)(xg + ((size_t)cb * T + t) * Gn + cg0);
        uint32_t laddr = lbase;
#pragma unroll
        for (int i = 0; i < 8; ++i) {
            asm volatile("global_load_async_to_lds_b128 %0, %1, off"
                         :: "v"(laddr), "v"(gaddr)
                         : "memory");
            laddr += 16;
            gaddr += 16;
        }
    };

    issue_stage(0); // prologue: stage xg[t=0]
    __syncthreads();

    const int m = mb * 16 + l15; // A-operand row (batch index) for this lane

    for (int t = 0; t < T; ++t) {
        // ------- phase 1: raw gates = h @ Whh^T  (no global traffic) -------
        v16bf Afrag[4];
#pragma unroll
        for (int kc = 0; kc < 4; ++kc) {
            const __bf16* hs = hstate + (size_t)m * Hn + kc * 32 + half * 16;
            v16bf av;
#pragma unroll
            for (int e = 0; e < 8; ++e) {
                av[e]     = hs[e];
                av[8 + e] = hs[8 + e];
            }
            Afrag[kc] = av;
        }

        if (t + 2 < T) // warm L2 two steps ahead of the async stage
            __builtin_prefetch(&xg[((size_t)cb * T + t + 2) * Gn + cg0], 0, 0);

#pragma unroll
        for (int j = 0; j < 4; ++j) {
            const int g = (ng * 4 + j) * 16 + l15;
            v8f acc;
#pragma unroll
            for (int e = 0; e < 8; ++e) acc[e] = 0.0f;
#pragma unroll
            for (int kc = 0; kc < 4; ++kc)
                acc = __builtin_amdgcn_wmma_f32_16x16x32_bf16(
                    false, Afrag[kc], false, Bfrag[j][kc], (short)0, acc,
                    false, false);
#pragma unroll
            for (int e = 0; e < 8; ++e) {
                const int b = mb * 16 + half * 8 + e;
                gates[(size_t)b * Gn + g] = acc[e];
            }
        }

        // xg[t] stage (issued last iteration / prologue) must be complete
        asm volatile("s_wait_asynccnt 0x0" ::: "memory");
        __syncthreads();

        // ------- phase 2: gate activation + c/h update ----------------------
#pragma unroll
        for (int r = 0; r < 8; ++r) {
            const int e = tid + r * 1024; // 0..8191 -> (b, j)
            const int b = e >> 7;
            const int j = e & 127;
            const size_t base = (size_t)b * Gn + j;
            const float ig = fast_sigmoid(gates[base]       + xstage[base]);
            const float fg = fast_sigmoid(gates[base + 128] + xstage[base + 128]);
            const float gg = fast_tanh   (gates[base + 256] + xstage[base + 256]);
            const float og = fast_sigmoid(gates[base + 384] + xstage[base + 384]);
            float c = fg * cstate[b * Hn + j] + ig * gg;
            cstate[b * Hn + j] = c;
            const float h = og * fast_tanh(c);
            hstate[b * Hn + j] = (__bf16)h;
            hout[(size_t)(b * T + t) * Hn + j] = h;
        }
        __syncthreads(); // xstage fully consumed; safe to overwrite

        if (t + 1 < T)
            issue_stage(t + 1); // overlaps next step's WMMA phase
    }
}

// ---------------------------------------------------------------------------
// Kernel 3: score[b,t] = sigmoid(sc[b,t,:] . linW + linb)
// ---------------------------------------------------------------------------
__global__ __launch_bounds__(256) void score_kernel(
    const float* __restrict__ sc, const float* __restrict__ linW,
    const float* __restrict__ linb, float* __restrict__ score, int N)
{
    const int i = blockIdx.x * blockDim.x + threadIdx.x;
    if (i >= N) return;
    const float* row = sc + (size_t)i * Hn;
    float acc = linb[0];
#pragma unroll 8
    for (int k = 0; k < Hn; ++k) acc = fmaf(row[k], linW[k], acc);
    score[i] = fast_sigmoid(acc);
}

// ---------------------------------------------------------------------------
// Kernel 4: select_feature. One block per batch row, 128 threads (1/channel).
// cumsum index is non-decreasing (score>0), so scatter uses register
// accumulation with flush-on-index-change; no atomics.
// ---------------------------------------------------------------------------
__global__ __launch_bounds__(128) void select_kernel(
    const float* __restrict__ feature, // [B,T,128]
    const float* __restrict__ score,   // [B,T]
    float* __restrict__ out)           // [B,792,128]
{
    __shared__ float sv[Tn];     // scaled score values
    __shared__ float cs[Tn];     // inclusive cumsum
    __shared__ float red[128];
    __shared__ float chunk[128];

    const int b = blockIdx.x, tid = threadIdx.x;

    // sum reduction
    float ps = 0.0f;
    for (int t = tid; t < Tn; t += 128) {
        const float v = score[(size_t)b * Tn + t];
        sv[t] = v; ps += v;
    }
    red[tid] = ps; __syncthreads();
    for (int off = 64; off; off >>= 1) {
        if (tid < off) red[tid] += red[tid + off];
        __syncthreads();
    }
    const float scale = (float)TOTAL_LEN / red[0];
    __syncthreads();

    // scale + max reduction
    float pm = 0.0f;
    for (int t = tid; t < Tn; t += 128) {
        const float v = sv[t] * scale;
        sv[t] = v; pm = fmaxf(pm, v);
    }
    red[tid] = pm; __syncthreads();
    for (int off = 64; off; off >>= 1) {
        if (tid < off) red[tid] = fmaxf(red[tid], red[tid + off]);
        __syncthreads();
    }
    const float mx = red[0];
    __syncthreads();
    if (mx >= 1.0f) {
        const float inv = 1.0f / mx;
        for (int t = tid; t < Tn; t += 128) sv[t] *= inv;
    }
    __syncthreads();

    // chunked inclusive scan
    const int CH = (Tn + 127) / 128; // 9
    const int t0 = tid * CH;
    float run = 0.0f;
    for (int k = 0; k < CH; ++k) {
        const int t = t0 + k;
        if (t < Tn) { run += sv[t]; cs[t] = run; }
    }
    chunk[tid] = run; __syncthreads();
    if (tid == 0) {
        float acc = 0.0f;
        for (int i = 0; i < 128; ++i) { const float v = chunk[i]; chunk[i] = acc; acc += v; }
    }
    __syncthreads();
    const float off0 = chunk[tid];
    for (int k = 0; k < CH; ++k) {
        const int t = t0 + k;
        if (t < Tn) cs[t] += off0;
    }
    __syncthreads();

    // zero this batch row's output
    for (int i = tid; i < TOTAL_LEN * Hn; i += 128)
        out[(size_t)b * TOTAL_LEN * Hn + i] = 0.0f;
    __syncthreads();

    // scatter: thread d handles channel d; idx sequence is thread-uniform
    const int d = tid;
    float accum = 0.0f;
    int cur = -1;
    for (int t = 0; t < Tn; ++t) {
        int idx = (int)floorf(cs[t]);
        idx = idx < 0 ? 0 : (idx > Tn - 1 ? Tn - 1 : idx);
        if (idx != cur) {
            if (cur >= 0 && cur < TOTAL_LEN)
                out[((size_t)b * TOTAL_LEN + cur) * Hn + d] += accum;
            accum = 0.0f; cur = idx;
        }
        accum += sv[t] * feature[((size_t)b * Tn + t) * Hn + d];
    }
    if (cur >= 0 && cur < TOTAL_LEN)
        out[((size_t)b * TOTAL_LEN + cur) * Hn + d] += accum;
}

// ---------------------------------------------------------------------------
// Host orchestration
// ---------------------------------------------------------------------------
extern "C" void kernel_launch(void* const* d_in, const int* in_sizes, int n_in,
                              void* d_out, int out_size, void* d_ws, size_t ws_size,
                              hipStream_t stream)
{
    (void)in_sizes; (void)n_in; (void)out_size; (void)ws_size;

    const int N = Bn * Tn; // 67584 rows

    const float* x = (const float*)d_in[0];
    const float* Wih[4] = { (const float*)d_in[1],  (const float*)d_in[5],
                            (const float*)d_in[9],  (const float*)d_in[13] };
    const float* Whh[4] = { (const float*)d_in[2],  (const float*)d_in[6],
                            (const float*)d_in[10], (const float*)d_in[14] };
    const float* bih[4] = { (const float*)d_in[3],  (const float*)d_in[7],
                            (const float*)d_in[11], (const float*)d_in[15] };
    const float* bhh[4] = { (const float*)d_in[4],  (const float*)d_in[8],
                            (const float*)d_in[12], (const float*)d_in[16] };
    const float* linW = (const float*)d_in[17];
    const float* linb = (const float*)d_in[18];

    // workspace layout
    char* ws = (char*)d_ws;
    float* xg    = (float*)ws;                                     // N*512 f32
    float* bufA  = (float*)(ws + (size_t)N * Gn * sizeof(float));  // N*128 f32
    float* bufB  = bufA + (size_t)N * Hn;                          // feature
    float* bufC  = bufB + (size_t)N * Hn;                          // sc
    float* score = bufC + (size_t)N * Hn;                          // N f32

    const float* lin[4]  = { x, bufA, bufB, bufA };
    float*       lout[4] = { bufA, bufB, bufA, bufC };

    const size_t ldsBytes = (size_t)Bn * Gn * 4   // xg stage f32
                          + (size_t)Bn * Gn * 4   // raw gates f32
                          + (size_t)Bn * Hn * 4   // c f32
                          + (size_t)Bn * Hn * 2;  // h bf16   -> 311296 B
    (void)hipFuncSetAttribute((const void*)lstm_recurrent_kernel,
                              hipFuncAttributeMaxDynamicSharedMemorySize,
                              (int)ldsBytes);

    const int tilesTotal = (N / 16) * (Gn / 16); // 4224 * 32 = 135168
    dim3 ggrid(tilesTotal / 8), gblk(256);

    for (int l = 0; l < 4; ++l) {
        gemm_xg_kernel<<<ggrid, gblk, 0, stream>>>(lin[l], Wih[l], bih[l], bhh[l], xg, N);
        lstm_recurrent_kernel<<<1, 1024, ldsBytes, stream>>>(xg, Whh[l], lout[l], Tn);
    }

    score_kernel<<<(N + 255) / 256, 256, 0, stream>>>(bufC, linW, linb, score, N);
    select_kernel<<<Bn, 128, 0, stream>>>(bufB, score, (float*)d_out);
}
